// CRFLoss_72378788873128
// MI455X (gfx1250) — matrix operand order repaired and checked by reading, not compiled, not run on hardware
//
#include <hip/hip_runtime.h>
#include <hip/hip_bf16.h>

// CRF loss, CDNA5 (gfx1250, wave32). Forward scan kept in scaled linear
// domain so the L x L logsumexp kernel becomes V x exp(T^T) — a dense GEMM
// done with v_wmma_f32_16x16x32_f16 (f16 in, f32 accumulate). Emission tiles
// are DMA'd per step by the Tensor Data Mover (tensor_load_to_lds), double
// buffered one scan iteration ahead, with TDM LDS padding (stride 66 dwords)
// for bank-conflict-free ds reads.

typedef __attribute__((ext_vector_type(16))) _Float16 v16h;
typedef __attribute__((ext_vector_type(8)))  float    v8f;
typedef __attribute__((ext_vector_type(4)))  unsigned int u32x4;
typedef __attribute__((ext_vector_type(8)))  int      i32x8;
typedef __attribute__((ext_vector_type(4)))  int      i32x4;

constexpr int B = 1024, T = 512, L = 64;
constexpr int UPITCH = 68;   // padded LDS row stride for U (floats)
constexpr int SPITCH = 66;   // TDM-padded row stride for emission tiles

#if __has_builtin(__builtin_amdgcn_tensor_load_to_lds)
#define USE_TDM 1
#else
#define USE_TDM 0
#endif

#if USE_TDM
__device__ __forceinline__ void tdm_issue_tile(const float* gaddr, unsigned lds_off) {
  // D# group 0: count=1 | lds_addr | 57-bit global addr | type=2
  const unsigned long long ga = (unsigned long long)(size_t)gaddr;
  u32x4 g0;
  g0[0] = 1u;                                          // count = 1 (valid)
  g0[1] = lds_off;                                     // lds_addr (bytes)
  g0[2] = (unsigned)ga;                                // global_addr[31:0]
  g0[3] = (unsigned)((ga >> 32) & 0x01FFFFFFu)         // global_addr[56:32]
        | 0x80000000u;                                 // type = 2 ("image")
  // D# group 1: data_size=4B, pad_enable, pad_interval=64dw, pad_amount=2dw,
  // tensor_dim0=64, tensor_dim1=16, tile_dim0=64, tile_dim1=16,
  // tensor_dim0_stride = T*L (elements between batch rows at fixed t)
  i32x8 g1;
  g1[0] = (int)((2u << 16) | (1u << 20) | (5u << 22) | (1u << 25));
  g1[1] = (int)(64u << 16);        // tensor_dim0[15:0] at bits [31:16]
  g1[2] = (int)(16u << 16);        // tensor_dim0 hi=0 ; tensor_dim1[15:0]
  g1[3] = (int)(64u << 16);        // tensor_dim1 hi=0 ; tile_dim0=64
  g1[4] = 16;                      // tile_dim1=16, tile_dim2=0 (2D)
  g1[5] = (int)(T * L);            // tensor_dim0_stride low32 = 32768
  g1[6] = 0;
  g1[7] = 0;
  i32x4 z4 = {};
#if defined(__clang_major__) && (__clang_major__ >= 23)
  i32x8 z8 = {};
  __builtin_amdgcn_tensor_load_to_lds(g0, g1, z4, z4, z8, 0);
#else
  __builtin_amdgcn_tensor_load_to_lds(g0, g1, z4, z4, 0);
#endif
}
#endif

// ---------------------------------------------------------------------------
// Kernel 1: forward normalizers. 1 workgroup (4 wave32) per 16 batches.
// Wave w owns output labels [16w, 16w+16).
// ---------------------------------------------------------------------------
__global__ __launch_bounds__(128) void crf_forward_kernel(
    const float* __restrict__ scores, const float* __restrict__ start,
    const float* __restrict__ Tmat,   const float* __restrict__ endv,
    float* __restrict__ nrm)
{
  __shared__ float Ush[16][UPITCH];
  __shared__ float Sbuf[2][16 * SPITCH];   // TDM double buffer (padded rows)
  __shared__ float Pred[16][8];
  __shared__ float Crow[16];
  __shared__ float Rinv[16];

  const int tid  = threadIdx.x;
  const int lane = tid & 31;
  const int w    = tid >> 5;      // wave id 0..3  -> N-tile
  const int hi   = lane >> 4;     // lane half
  const int nloc = lane & 15;     // column within 16x16 tile
  const int bg   = blockIdx.x * 16;

  // ---- E = exp(Tmat^T) in WMMA B-operand layout, f16, resident in regs ----
  v16h e0, e1;
  {
    const int j = 16 * w + nloc;                 // 'to' label
#pragma unroll
    for (int v = 0; v < 8; ++v) {
#pragma unroll
      for (int h = 0; h < 2; ++h) {
        const int kb = 2 * v + h + 16 * hi;      // 'from' label (K)
        e0[2 * v + h] = (_Float16)__expf(Tmat[j * L + kb]);
        e1[2 * v + h] = (_Float16)__expf(Tmat[j * L + 32 + kb]);
      }
    }
  }

  // ---- init (t = 0): u = exp(start + scores[:,0,:]) ----
  if (tid < 16) Crow[tid] = 0.0f;
#pragma unroll
  for (int p = 0; p < 8; ++p) {
    const int e = tid + 128 * p;                 // coalesced over 16x64 tile
    const int m = e >> 6, j = e & 63;
    Ush[m][j] = __expf(start[j] + scores[(size_t)(bg + m) * (T * L) + j]);
  }
#if USE_TDM
  // Prologue: DMA the t=1 emission tile into buffer 1.
  if (tid == 0)
    tdm_issue_tile(scores + (size_t)bg * (T * L) + (size_t)1 * L,
                   (unsigned)(size_t)&Sbuf[1][0]);
#endif
  __syncthreads();
  {
    const int r = tid & 15, c = tid >> 4;
    float mx = Ush[r][8 * c];
#pragma unroll
    for (int q = 1; q < 8; ++q) mx = fmaxf(mx, Ush[r][8 * c + q]);
    Pred[r][c] = mx;
  }
  __syncthreads();
  if (tid < 16) {
    float q = Pred[tid][0];
#pragma unroll
    for (int c = 1; c < 8; ++c) q = fmaxf(q, Pred[tid][c]);
    Crow[tid] += __logf(q);
    Rinv[tid] = 1.0f / q;
  }
  __syncthreads();

  const int mA = lane & 15;                      // A-operand row
#if !USE_TDM
  // fallback: per-lane global emission loads in D layout
  const float* sbase = scores + (size_t)(bg + 8 * hi) * (T * L) + 16 * w + nloc;
#endif

  for (int t = 1; t < T; ++t) {
#if USE_TDM
    if (tid == 0) {
      __builtin_amdgcn_s_wait_tensorcnt(0);      // tile for t has landed
      if (t + 1 < T)                             // prefetch tile for t+1
        tdm_issue_tile(scores + (size_t)bg * (T * L) + (size_t)(t + 1) * L,
                       (unsigned)(size_t)&Sbuf[(t + 1) & 1][0]);
    }
#else
    const float* sp = sbase + (size_t)t * L;
    float sv[8];
#pragma unroll
    for (int v = 0; v < 8; ++v) sv[v] = sp[v * (T * L)];
#endif

    // ---- A operand: V = U * (1/q), f16, A 16-bit 16x32 layout ----
    const float rinv = Rinv[mA];
    v16h a0, a1;
#pragma unroll
    for (int v = 0; v < 8; ++v) {
      const int k0 = 2 * v + 16 * hi;  // K within block
      const float2 p0 = *(const float2*)&Ush[mA][k0];        // K half 0..31
      const float2 p1 = *(const float2*)&Ush[mA][32 + k0];   // K half 32..63
      a0[2 * v]     = (_Float16)(p0.x * rinv);
      a0[2 * v + 1] = (_Float16)(p0.y * rinv);
      a1[2 * v]     = (_Float16)(p1.x * rinv);
      a1[2 * v + 1] = (_Float16)(p1.y * rinv);
    }
    __syncthreads();   // all reads of U done; TDM tile for t published to WG

    // ---- S = V x E : two chained WMMAs cover K = 64 ----
    v8f acc = {};
    acc = __builtin_amdgcn_wmma_f32_16x16x32_f16(false, a0, false, e0,
                                                 (short)0, acc, false, false);
    acc = __builtin_amdgcn_wmma_f32_16x16x32_f16(false, a1, false, e1,
                                                 (short)0, acc, false, false);

#if USE_TDM
    // emission values from the TDM-padded LDS tile (D layout: m = v + 8*hi)
    const float* sb = &Sbuf[t & 1][(8 * hi) * SPITCH + 16 * w + nloc];
    float sv[8];
#pragma unroll
    for (int v = 0; v < 8; ++v) sv[v] = sb[v * SPITCH];
#endif

    // ---- u = exp(s_t) * S ; store (D layout -> LDS rows) ----
#pragma unroll
    for (int v = 0; v < 8; ++v)
      Ush[v + 8 * hi][16 * w + nloc] = __expf(sv[v]) * acc[v];
    __syncthreads();

    // ---- q = rowmax(u); C += log q; Rinv = 1/q ----
    {
      const int r = tid & 15, c = tid >> 4;
      float mx = Ush[r][8 * c];
#pragma unroll
      for (int q = 1; q < 8; ++q) mx = fmaxf(mx, Ush[r][8 * c + q]);
      Pred[r][c] = mx;
    }
    __syncthreads();
    if (tid < 16) {
      float q = Pred[tid][0];
#pragma unroll
      for (int c = 1; c < 8; ++c) q = fmaxf(q, Pred[tid][c]);
      Crow[tid] += __logf(q);
      Rinv[tid] = 1.0f / q;
    }
    __syncthreads();
  }

  // ---- normalizer = C + log( (1/q) * sum_j u_j * exp(end_j) ) ----
  {
    const int r = tid & 15, c = tid >> 4;
    float ps = 0.0f;
#pragma unroll
    for (int q = 0; q < 8; ++q)
      ps += Ush[r][8 * c + q] * __expf(endv[8 * c + q]);
    Pred[r][c] = ps;
  }
  __syncthreads();
  if (tid < 16) {
    float s = 0.0f;
#pragma unroll
    for (int c = 0; c < 8; ++c) s += Pred[tid][c];
    nrm[bg + tid] = Crow[tid] + __logf(Rinv[tid] * s);
  }
}

// ---------------------------------------------------------------------------
// Kernel 2: gold-path score per batch (pure gather-sum, one thread per chain)
// ---------------------------------------------------------------------------
__global__ __launch_bounds__(256) void crf_target_kernel(
    const float* __restrict__ scores, const int* __restrict__ targets,
    const float* __restrict__ start,  const float* __restrict__ Tmat,
    const float* __restrict__ endv,   float* __restrict__ ts)
{
  const int b = blockIdx.x * blockDim.x + threadIdx.x;
  if (b >= B) return;
  const int*   tg = targets + (size_t)b * T;
  const float* sc = scores + (size_t)b * T * L;
  int yp = tg[0];
  float acc = start[yp] + sc[yp];
  for (int t = 1; t < T; ++t) {
    const int y = tg[t];
    acc += sc[t * L + y] + Tmat[y * L + yp];
    yp = y;
  }
  ts[b] = acc + endv[yp];
}

// ---------------------------------------------------------------------------
// Kernel 3: deterministic mean reduction (single block, no atomics)
// ---------------------------------------------------------------------------
__global__ __launch_bounds__(256) void crf_reduce_kernel(
    const float* __restrict__ nrm, const float* __restrict__ ts,
    float* __restrict__ out)
{
  __shared__ float sm[256];
  const int tid = threadIdx.x;
  float s = 0.0f;
  for (int b = tid; b < B; b += 256) s += nrm[b] - ts[b];
  sm[tid] = s;
  __syncthreads();
  for (int off = 128; off > 0; off >>= 1) {
    if (tid < off) sm[tid] += sm[tid + off];
    __syncthreads();
  }
  if (tid == 0) out[0] = sm[0] * (1.0f / (float)B);
}

extern "C" void kernel_launch(void* const* d_in, const int* in_sizes, int n_in,
                              void* d_out, int out_size, void* d_ws, size_t ws_size,
                              hipStream_t stream) {
  (void)in_sizes; (void)n_in; (void)out_size; (void)ws_size;
  const float* scores  = (const float*)d_in[0];   // [B,T,L] f32
  const int*   targets = (const int*)  d_in[1];   // [B,T]   i32
  const float* start   = (const float*)d_in[2];   // [L]     f32
  const float* Tmat    = (const float*)d_in[3];   // [L,L]   f32 (T[to,from])
  const float* endv    = (const float*)d_in[4];   // [L]     f32
  float* out = (float*)d_out;
  float* nrm = (float*)d_ws;                      // [B]
  float* ts  = nrm + B;                           // [B]

  crf_forward_kernel<<<B / 16, 128, 0, stream>>>(scores, start, Tmat, endv, nrm);
  crf_target_kernel<<<B / 256, 256, 0, stream>>>(scores, targets, start, Tmat, endv, ts);
  crf_reduce_kernel<<<1, 256, 0, stream>>>(nrm, ts, out);
}
